// MinkFPN_17703855194358
// MI455X (gfx1250) — matrix operand order repaired and checked by reading, not compiled, not run on hardware
//
#include <hip/hip_runtime.h>
#include <hip/hip_bf16.h>

typedef __attribute__((ext_vector_type(16))) _Float16 v16h;
typedef __attribute__((ext_vector_type(8)))  _Float16 v8h;
typedef __attribute__((ext_vector_type(4)))  _Float16 v4h;
typedef __attribute__((ext_vector_type(8)))  float    v8f;

// ---------------------------------------------------------------------------
// Weight swizzle: f32 [K][CIN][NCOL] -> f16 fragment order
// [k][chunk(CIN/32)][slab(NCOL/16)][lane(32)][t(16)]
// so each lane's B fragment (kb = chunk*32 + half*8 + (t&7) + (t>>3)*16, col = slab*16 + ln)
// is one contiguous 32-byte run.
// ---------------------------------------------------------------------------
__global__ void swizzle_w(const float* __restrict__ w, _Float16* __restrict__ d,
                          int K, int CIN, int NCOL) {
    const int per_k = CIN * NCOL;
    const int per_chunk = NCOL * 32;                 // halfs per (k,chunk)
    long i = (long)blockIdx.x * 256 + threadIdx.x;
    if (i >= (long)K * per_k) return;
    int k = (int)(i / per_k);
    int r = (int)(i - (long)k * per_k);
    int chunk = r / per_chunk;  r -= chunk * per_chunk;
    int slab  = r >> 9;         r &= 511;            // 32 lanes * 16 t
    int lane  = r >> 4;
    int t     = r & 15;
    int half = lane >> 4, ln = lane & 15;
    int kb  = (chunk << 5) + (half << 3) + (t & 7) + ((t >> 3) << 4);
    int col = (slab << 4) + ln;
    d[i] = (_Float16)w[(size_t)k * per_k + (size_t)kb * NCOL + col];
}

// ---------------------------------------------------------------------------
// conv0: 125 taps, cin=1, cout=64. out[oo] += x[ii] * W[k][0][:]
// ---------------------------------------------------------------------------
__global__ void conv0_tap(const float* __restrict__ x, const float* __restrict__ w,
                          const int* __restrict__ mi, const int* __restrict__ mo,
                          int P, float* __restrict__ acc) {
    int k = blockIdx.y;
    int idx = blockIdx.x * 256 + threadIdx.x;
    int p = idx >> 6, c = idx & 63;
    if (p >= P) return;
    int ii = mi[(size_t)k * P + p];
    int oo = mo[(size_t)k * P + p];
    atomicAdd(&acc[(size_t)oo * 64 + c], x[ii] * w[k * 64 + c]);
}

// ---------------------------------------------------------------------------
// Sparse conv via WMMA.  64-row M supertile per block, 4 waves; wave w owns
// output columns [16w,16w+16) across 4 M-tiles, so each B fragment (one
// 32-byte contiguous load from the swizzled weights) feeds 4 WMMAs.
// A fragments come from the f16 LDS tile as two 16-byte DS vector loads.
// CIN==128 gathers channels 0-63 from xa and 64-127 from xb (fused concat).
// Scatter is a branchless atomic: padding / out-of-tile rows select a trash
// row via address cndmask (extends the reference's dummy-row trick).
// ---------------------------------------------------------------------------
template <int CIN>
__global__ void sparse_conv_wmma(const float* __restrict__ xa, const float* __restrict__ xb,
                                 const _Float16* __restrict__ wsw,
                                 const int* __restrict__ mi, const int* __restrict__ mo,
                                 int P, float* __restrict__ acc, float* __restrict__ trash)
{
    constexpr int NCH = CIN / 32;                    // k-chunks
    __shared__ _Float16 sA[64 * CIN];
    __shared__ int s_ii[64];
    __shared__ int s_oo[64];

    const int k   = blockIdx.y;
    const int m0  = blockIdx.x * 64;
    const int tid = threadIdx.x;

    if (tid < 64) {
        int row = m0 + tid;
        int ii = 0, oo = -1;
        if (row < P) { ii = mi[(size_t)k * P + row]; oo = mo[(size_t)k * P + row]; }
        s_ii[tid] = ii;
        s_oo[tid] = oo;
    }
    __syncthreads();

    // Gather + f32->f16 convert, float4-vectorized: 64 rows x CIN channels.
    constexpr int QPR = CIN / 4;                     // float4 quads per row
    for (int i = tid; i < 64 * QPR; i += 128) {
        int r  = i / QPR;
        int c4 = (i - r * QPR) << 2;
        const float* src;
        if constexpr (CIN == 128)
            src = (c4 < 64) ? xa + (size_t)s_ii[r] * 64 + c4
                            : xb + (size_t)s_ii[r] * 64 + (c4 - 64);
        else
            src = xa + (size_t)s_ii[r] * 64 + c4;
        float4 v = *(const float4*)src;
        v4h h = { (_Float16)v.x, (_Float16)v.y, (_Float16)v.z, (_Float16)v.w };
        *(v4h*)(sA + r * CIN + c4) = h;
    }
    __syncthreads();

    const int lane = tid & 31;
    const int wave = tid >> 5;      // 0..3 -> 16-col slab
    const int half = lane >> 4;
    const int ln   = lane & 15;

    const v16h* wb = (const v16h*)wsw + ((size_t)k * NCH) * 128 + (wave << 5) + lane;
    __builtin_prefetch(wb, 0, 1);   // global_prefetch_b8

    v8f cf[4] = {};
#pragma unroll
    for (int chunk = 0; chunk < NCH; ++chunk) {
        v16h b = wb[chunk * 128];
#pragma unroll
        for (int mt = 0; mt < 4; ++mt) {
            const _Float16* ab = sA + ((mt << 4) + ln) * CIN + (chunk << 5) + (half << 3);
            v8h lo = *(const v8h*)ab;            // A-frag K 0..7   (this half)
            v8h hi = *(const v8h*)(ab + 16);     // A-frag K 16..23 (this half)
            v16h a = __builtin_shufflevector(lo, hi, 0, 1, 2, 3, 4, 5, 6, 7,
                                             8, 9, 10, 11, 12, 13, 14, 15);
            cf[mt] = __builtin_amdgcn_wmma_f32_16x16x32_f16(false, a, false, b,
                                                            (short)0, cf[mt],
                                                            false, false);
        }
    }

    // Branchless scatter-add: C/D VGPR r -> M = r + 8*half, column = ln.
#pragma unroll
    for (int mt = 0; mt < 4; ++mt) {
#pragma unroll
        for (int r = 0; r < 8; ++r) {
            int oo = s_oo[(mt << 4) + r + (half << 3)];
            float* dst = (oo >= 0) ? acc + (size_t)oo * 64 : trash;
            atomicAdd(dst + (wave << 4) + ln, cf[mt][r]);
        }
    }
}

// ---------------------------------------------------------------------------
// BatchNorm (training-mode stats) + ReLU
// ---------------------------------------------------------------------------
__global__ void bn_stats(const float* __restrict__ x, int n, float* __restrict__ stats) {
    __shared__ float ssum[256];
    __shared__ float ssq[256];
    int c = blockIdx.x, t = threadIdx.x;
    float s = 0.f, q = 0.f;
    for (int i = t; i < n; i += 256) {
        float v = x[(size_t)i * 64 + c];
        s += v; q += v * v;
    }
    ssum[t] = s; ssq[t] = q;
    __syncthreads();
    for (int o = 128; o > 0; o >>= 1) {
        if (t < o) { ssum[t] += ssum[t + o]; ssq[t] += ssq[t + o]; }
        __syncthreads();
    }
    if (t == 0) {
        float mu  = ssum[0] / (float)n;
        float var = ssq[0] / (float)n - mu * mu;
        stats[c]      = mu;
        stats[64 + c] = rsqrtf(var + 1e-5f);
    }
}

__global__ void bn_relu(float* __restrict__ x, int n, const float* __restrict__ stats,
                        const float* __restrict__ g, const float* __restrict__ b) {
    int i = blockIdx.x * 256 + threadIdx.x;
    if (i >= n * 64) return;
    int c = i & 63;
    float v = (x[i] - stats[c]) * stats[64 + c] * g[c] + b[c];
    x[i] = v > 0.f ? v : 0.f;
}

// ---------------------------------------------------------------------------
// Final dense GEMM: [n0,64] @ Wf[64,256], swizzled Wf.  16 waves/block, each
// owns a 16-column slab.  Branchless tail: overflow rows store to trash.
// ---------------------------------------------------------------------------
__global__ void final_gemm_wmma(const float* __restrict__ x, const _Float16* __restrict__ wf,
                                int n0, float* __restrict__ out, float* __restrict__ trash) {
    __shared__ _Float16 sA[16 * 64];
    int m0 = blockIdx.x * 16;
    int tid = threadIdx.x;
    for (int i = tid; i < 16 * 16; i += 512) {       // 16 rows * 16 quads
        int r  = i >> 4;
        int c4 = (i & 15) << 2;
        int row = m0 + r;
        if (row >= n0) row = n0 - 1;
        float4 v = *(const float4*)(x + (size_t)row * 64 + c4);
        v4h h = { (_Float16)v.x, (_Float16)v.y, (_Float16)v.z, (_Float16)v.w };
        *(v4h*)(sA + r * 64 + c4) = h;
    }
    __syncthreads();

    int lane = tid & 31, wave = tid >> 5, half = lane >> 4, ln = lane & 15;
    const v16h* wb = (const v16h*)wf + (wave << 5) + lane;   // [chunk][slab16][lane][16]
    v8f cf = {};
#pragma unroll
    for (int chunk = 0; chunk < 2; ++chunk) {
        v16h b = wb[chunk * 512];                    // 16 slabs * 32 lanes per chunk
        const _Float16* ab = sA + ln * 64 + (chunk << 5) + (half << 3);
        v8h lo = *(const v8h*)ab;
        v8h hi = *(const v8h*)(ab + 16);
        v16h a = __builtin_shufflevector(lo, hi, 0, 1, 2, 3, 4, 5, 6, 7,
                                         8, 9, 10, 11, 12, 13, 14, 15);
        cf = __builtin_amdgcn_wmma_f32_16x16x32_f16(false, a, false, b, (short)0, cf,
                                                    false, false);
    }
#pragma unroll
    for (int r = 0; r < 8; ++r) {
        int row = m0 + r + (half << 3);
        float* dst = (row < n0) ? out + (size_t)row * 256 : trash;
        dst[(wave << 4) + ln] = cf[r];
    }
}

// ---------------------------------------------------------------------------
// Host orchestration
// ---------------------------------------------------------------------------
static inline unsigned cdiv(unsigned a, unsigned b) { return (a + b - 1) / b; }

extern "C" void kernel_launch(void* const* d_in, const int* in_sizes, int n_in,
                              void* d_out, int out_size, void* d_ws, size_t ws_size,
                              hipStream_t stream) {
    (void)n_in; (void)out_size; (void)ws_size;
    // ---- input indices (setup_inputs dict order, depth-first) ----
    const float* feats = (const float*)d_in[0];
    const float* Wc0 = (const float*)d_in[1];
    const float* g0 = (const float*)d_in[2];  const float* b0 = (const float*)d_in[3];
    const float* We0a = (const float*)d_in[4];
    const float* ge0a = (const float*)d_in[5]; const float* be0a = (const float*)d_in[6];
    const float* We0b = (const float*)d_in[7];
    const float* ge0b = (const float*)d_in[8]; const float* be0b = (const float*)d_in[9];
    const float* Wp0 = (const float*)d_in[10];
    const float* We1a = (const float*)d_in[11];
    const float* ge1a = (const float*)d_in[12]; const float* be1a = (const float*)d_in[13];
    const float* We1b = (const float*)d_in[14];
    const float* ge1b = (const float*)d_in[15]; const float* be1b = (const float*)d_in[16];
    const float* Wp1 = (const float*)d_in[17];
    const float* Wu0 = (const float*)d_in[18];
    const float* Wd0a = (const float*)d_in[19];
    const float* gd0a = (const float*)d_in[20]; const float* bd0a = (const float*)d_in[21];
    const float* Wd0b = (const float*)d_in[22];
    const float* gd0b = (const float*)d_in[23]; const float* bd0b = (const float*)d_in[24];
    const float* Wu1 = (const float*)d_in[25];
    const float* Wd1a = (const float*)d_in[26];
    const float* gd1a = (const float*)d_in[27]; const float* bd1a = (const float*)d_in[28];
    const float* Wd1b = (const float*)d_in[29];
    const float* gd1b = (const float*)d_in[30]; const float* bd1b = (const float*)d_in[31];
    const float* Wf = (const float*)d_in[32];
    const int* c0_in = (const int*)d_in[33]; const int* c0_out = (const int*)d_in[34];
    const int* e0_in = (const int*)d_in[35]; const int* e0_out = (const int*)d_in[36];
    const int* e1_in = (const int*)d_in[37]; const int* e1_out = (const int*)d_in[38];
    const int* p0_in = (const int*)d_in[39]; const int* p0_out = (const int*)d_in[40];
    const int* p1_in = (const int*)d_in[41]; const int* p1_out = (const int*)d_in[42];
    const int* u0_in = (const int*)d_in[43]; const int* u0_out = (const int*)d_in[44];
    const int* u1_in = (const int*)d_in[45]; const int* u1_out = (const int*)d_in[46];

    // ---- host-derivable sizes ----
    const int n0  = in_sizes[0];            // feats is [n0,1]
    const int Pc0 = in_sizes[33] / 125;     // == n0 (center tap always hits)
    const int Pe0 = in_sizes[35] / 27;      // == n0
    const int n1  = in_sizes[37] / 27;      // center tap of L1 same-level map
    const int Pp0 = in_sizes[39] / 8;
    const int Pp1 = in_sizes[41] / 8;
    const int Pu0 = in_sizes[43] / 8;
    const int Pu1 = in_sizes[45] / 8;

    // ---- workspace carve-up (256B aligned) ----
    char* cur = (char*)d_ws;
    auto alloc = [&](size_t bytes) {
        char* p = cur;
        cur += (bytes + 255) & ~(size_t)255;
        return (void*)p;
    };
    _Float16* hWe0a = (_Float16*)alloc(27 * 64 * 64 * 2);
    _Float16* hWe0b = (_Float16*)alloc(27 * 64 * 64 * 2);
    _Float16* hWe1a = (_Float16*)alloc(27 * 64 * 64 * 2);
    _Float16* hWe1b = (_Float16*)alloc(27 * 64 * 64 * 2);
    _Float16* hWd0a = (_Float16*)alloc(27 * 128 * 64 * 2);
    _Float16* hWd0b = (_Float16*)alloc(27 * 64 * 64 * 2);
    _Float16* hWd1a = (_Float16*)alloc(27 * 128 * 64 * 2);
    _Float16* hWd1b = (_Float16*)alloc(27 * 64 * 64 * 2);
    _Float16* hWp0  = (_Float16*)alloc(8 * 64 * 64 * 2);
    _Float16* hWp1  = (_Float16*)alloc(8 * 64 * 64 * 2);
    _Float16* hWu0  = (_Float16*)alloc(8 * 64 * 64 * 2);
    _Float16* hWu1  = (_Float16*)alloc(8 * 64 * 64 * 2);
    _Float16* hWf   = (_Float16*)alloc(64 * 256 * 2);
    const size_t szL0 = (size_t)(n0 + 1) * 64 * 4;
    const size_t szL1 = (size_t)(n1 + 1) * 64 * 4;
    float* A     = (float*)alloc(szL0);          // L0 buffer 1
    float* Bb    = (float*)alloc(szL0);          // L0 buffer 2
    float* Cc    = (float*)alloc(szL0);          // L0 buffer 3 (decoder up-path)
    float* P1a   = (float*)alloc(szL1);          // L1 buffer 1
    float* P1b   = (float*)alloc(szL1);          // L1 buffer 2
    float* P1c   = (float*)alloc(szL1);          // L1 buffer 3 (decoder up-path)
    float* X2    = (float*)alloc(szL1);          // L2 buffer (n2+1 <= n1+1 rows)
    float* stats = (float*)alloc(128 * 4);
    float* trash = (float*)alloc(256 * 4);

    // ---- weight swizzle (f32 -> fragment-ordered f16) ----
    auto swz = [&](const float* s, _Float16* d, int K, int CIN, int NCOL) {
        int n = K * CIN * NCOL;
        swizzle_w<<<cdiv(n, 256), 256, 0, stream>>>(s, d, K, CIN, NCOL);
    };
    swz(We0a, hWe0a, 27, 64, 64);  swz(We0b, hWe0b, 27, 64, 64);
    swz(We1a, hWe1a, 27, 64, 64);  swz(We1b, hWe1b, 27, 64, 64);
    swz(Wd0a, hWd0a, 27, 128, 64); swz(Wd0b, hWd0b, 27, 64, 64);
    swz(Wd1a, hWd1a, 27, 128, 64); swz(Wd1b, hWd1b, 27, 64, 64);
    swz(Wp0, hWp0, 8, 64, 64);     swz(Wp1, hWp1, 8, 64, 64);
    swz(Wu0, hWu0, 8, 64, 64);     swz(Wu1, hWu1, 8, 64, 64);
    swz(Wf, hWf, 1, 64, 256);

    auto conv64 = [&](const float* x, const _Float16* w, const int* mi, const int* mo,
                      int P, int K, float* acc, size_t accBytes) {
        hipMemsetAsync(acc, 0, accBytes, stream);
        sparse_conv_wmma<64><<<dim3(cdiv(P, 64), K), 128, 0, stream>>>(
            x, nullptr, w, mi, mo, P, acc, trash);
    };
    auto conv128 = [&](const float* xa, const float* xb, const _Float16* w,
                       const int* mi, const int* mo, int P, int K,
                       float* acc, size_t accBytes) {
        hipMemsetAsync(acc, 0, accBytes, stream);
        sparse_conv_wmma<128><<<dim3(cdiv(P, 64), K), 128, 0, stream>>>(
            xa, xb, w, mi, mo, P, acc, trash);
    };
    auto bn = [&](float* x, int n, const float* g, const float* b) {
        bn_stats<<<64, 256, 0, stream>>>(x, n, stats);
        bn_relu<<<cdiv((unsigned)n * 64, 256), 256, 0, stream>>>(x, n, stats, g, b);
    };

    // ---- forward pass ----
    // conv0 (k=5^3, cin=1) -> A ; BN0
    hipMemsetAsync(A, 0, szL0, stream);
    conv0_tap<<<dim3(cdiv((unsigned)Pc0 * 64, 256), 125), 256, 0, stream>>>(
        feats, Wc0, c0_in, c0_out, Pc0, A);
    bn(A, n0, g0, b0);
    // encoder block L0: e0 = BN(conv(BN(conv(A))))
    conv64(A, hWe0a, e0_in, e0_out, Pe0, 27, Bb, szL0);  bn(Bb, n0, ge0a, be0a);
    conv64(Bb, hWe0b, e0_in, e0_out, Pe0, 27, A, szL0);  bn(A, n0, ge0b, be0b);   // A = e0
    // pool L0->L1
    conv64(A, hWp0, p0_in, p0_out, Pp0, 8, P1a, szL1);
    // encoder block L1
    conv64(P1a, hWe1a, e1_in, e1_out, n1, 27, P1b, szL1); bn(P1b, n1, ge1a, be1a);
    conv64(P1b, hWe1b, e1_in, e1_out, n1, 27, P1a, szL1); bn(P1a, n1, ge1b, be1b); // P1a = e1
    // pool L1->L2
    conv64(P1a, hWp1, p1_in, p1_out, Pp1, 8, X2, szL1);
    // up L2->L1; decoder block L1 reads cat([up, e1]) directly (fused concat)
    conv64(X2, hWu0, u0_in, u0_out, Pu0, 8, P1c, szL1);
    conv128(P1c, P1a, hWd0a, e1_in, e1_out, n1, 27, P1b, szL1); bn(P1b, n1, gd0a, bd0a);
    conv64(P1b, hWd0b, e1_in, e1_out, n1, 27, P1a, szL1);       bn(P1a, n1, gd0b, bd0b);
    // up L1->L0; decoder block L0 reads cat([up, e0]) directly
    conv64(P1a, hWu1, u1_in, u1_out, Pu1, 8, Cc, szL0);
    conv128(Cc, A, hWd1a, e0_in, e0_out, Pe0, 27, Bb, szL0);    bn(Bb, n0, gd1a, bd1a);
    conv64(Bb, hWd1b, e0_in, e0_out, Pe0, 27, A, szL0);         bn(A, n0, gd1b, bd1b);
    // final 1x1 -> [n0, 256]
    final_gemm_wmma<<<cdiv(n0, 16), 512, 0, stream>>>(A, hWf, n0, (float*)d_out, trash);
}
